// FixedFMoE_79285096284552
// MI455X (gfx1250) — matrix operand order repaired and checked by reading, not compiled, not run on hardware
//
#include <hip/hip_runtime.h>
#include <hip/hip_bf16.h>
#include <math.h>

#define N_TOK 8192
#define D     1024
#define E     16
#define TOPK  2
#define NLAB  2
#define LN_EPS 1e-5f
#define TILE_M 64
#define TILES_PER_E (N_TOK / TILE_M)   // 128

typedef __bf16 v16bf __attribute__((ext_vector_type(16)));
typedef __bf16 v8bf  __attribute__((ext_vector_type(8)));
typedef float  v8f   __attribute__((ext_vector_type(8)));

// ---------------------------------------------------------------- gate + top2
__global__ void gate_topk_kernel(const float* __restrict__ x,
                                 const float* __restrict__ gw,
                                 const float* __restrict__ gb,
                                 int* __restrict__ idx,
                                 int* __restrict__ counts) {
  __shared__ float lg[16][E];
  int tid = threadIdx.x;
  int lt  = tid >> 4;          // local token 0..15
  int e   = tid & 15;          // expert 0..15
  int token = blockIdx.x * 16 + lt;
  const float* xr = x + (size_t)token * D;
  float acc = gb[e];
  for (int d = 0; d < D; ++d) acc += xr[d] * gw[d * E + e];
  lg[lt][e] = acc;
  __syncthreads();
  if (e == 0) {
    int i0 = 0; float v0 = lg[lt][0];
    for (int j = 1; j < E; ++j) { float v = lg[lt][j]; if (v > v0) { v0 = v; i0 = j; } }
    int i1 = (i0 == 0) ? 1 : 0; float v1 = lg[lt][i1];
    for (int j = 0; j < E; ++j) {
      if (j == i0) continue;
      float v = lg[lt][j];
      if (v > v1) { v1 = v; i1 = j; }
    }
    idx[token * TOPK + 0] = i0;
    idx[token * TOPK + 1] = i1;
    atomicAdd(&counts[i0], 1);
    atomicAdd(&counts[i1], 1);
  }
}

// ---------------------------------------------------------------- 16-wide scan
__global__ void scan_kernel(const int* __restrict__ counts,
                            int* __restrict__ offsets,
                            int* __restrict__ cursor) {
  if (threadIdx.x == 0) {
    int off = 0;
    for (int e = 0; e < E; ++e) { offsets[e] = off; off += counts[e]; }
    offsets[E] = off;
  }
  if (threadIdx.x < E) cursor[threadIdx.x] = 0;
}

// ---------------------------------------------------------------- scatter slots
__global__ void scatter_kernel(const int* __restrict__ idx,
                               const int* __restrict__ offsets,
                               int* __restrict__ cursor,
                               int* __restrict__ slotlist) {
  int token = blockIdx.x * blockDim.x + threadIdx.x;
  if (token >= N_TOK) return;
  for (int k = 0; k < TOPK; ++k) {
    int e = idx[token * TOPK + k];
    int pos = atomicAdd(&cursor[e], 1);
    slotlist[offsets[e] + pos] = token * TOPK + k;
  }
}

// ------------------------------------------- W[e][k][n] f32 -> Wt[e][n][k] bf16
__global__ void transpose_cvt_kernel(const float* __restrict__ W,
                                     __bf16* __restrict__ Wt) {
  __shared__ float tile[32][33];
  int e  = blockIdx.z;
  int kb = blockIdx.x * 32;
  int nb = blockIdx.y * 32;
  const float* Wg  = W  + (size_t)e * D * D;
  __bf16*      Wtg = Wt + (size_t)e * D * D;
  for (int r = threadIdx.y; r < 32; r += 8)
    tile[r][threadIdx.x] = Wg[(size_t)(kb + r) * D + nb + threadIdx.x];
  __syncthreads();
  for (int r = threadIdx.y; r < 32; r += 8)
    Wtg[(size_t)(nb + r) * D + kb + threadIdx.x] = (__bf16)tile[threadIdx.x][r];
}

// --------------------------------------------- WMMA fragment helpers (bf16)
// B 32x16 (KxN): lanes 0-15 -> N=lane, K=0..15 ; lanes 16-31 -> N=lane-16, K=16..31
__device__ __forceinline__ v16bf load_bfragB_f32(const float* __restrict__ Wg,
                                                 int kb, int col, int lhalf) {
  v16bf b;
  int k0 = kb + lhalf * 16;
#pragma unroll
  for (int j = 0; j < 16; ++j) b[j] = (__bf16)Wg[(size_t)(k0 + j) * D + col];
  return b;
}
// pre-transposed bf16 weights: contiguous 32B per lane -> 2x global_load_b128
__device__ __forceinline__ v16bf load_bfragB_bt(const __bf16* __restrict__ Wt,
                                                int kb, int col, int lhalf) {
  int k0 = kb + lhalf * 16;
  return *(const v16bf*)(Wt + (size_t)col * D + k0);
}
// A 16x32 (MxK): lane half selects K base 0 / 16; elems 0..15 -> K+0..15
__device__ __forceinline__ v16bf load_afragLDS(const __bf16* base) {
  v8bf lo = *(const v8bf*)(base);
  v8bf hi = *(const v8bf*)(base + 8);
  v16bf a;
#pragma unroll
  for (int j = 0; j < 8; ++j) { a[j] = lo[j]; a[j + 8] = hi[j]; }
  return a;
}
__device__ __forceinline__ float gelu_exact(float v) {
  return 0.5f * v * (1.0f + erff(v * 0.70710678118654752f));
}

// ---------------------------------------------------------------- expert GEMMs
template <bool PRE>
__launch_bounds__(256, 1)
__global__ void expert_kernel(const float* __restrict__ x,
                              const float* __restrict__ W1, const float* __restrict__ b1,
                              const float* __restrict__ W2, const float* __restrict__ b2,
                              const __bf16* __restrict__ Wt1,
                              const __bf16* __restrict__ Wt2,
                              const float* __restrict__ lns, const float* __restrict__ lnb,
                              const float* __restrict__ cw,  const float* __restrict__ cb,
                              const int* __restrict__ offsets,
                              const int* __restrict__ slotlist,
                              float* __restrict__ moe_out,
                              float* __restrict__ conf_out) {
  extern __shared__ char smem_raw[];
  __bf16* xb    = (__bf16*)smem_raw;              // TILE_M x D bf16 (128 KB)
  __bf16* hb    = xb + TILE_M * D;                // TILE_M x D bf16 (128 KB)
  float*  scw   = (float*)(hb + TILE_M * D);      // D floats: scale*conf_w
  int*    slots = (int*)(scw + D);                // TILE_M slot ids
  float*  red   = (float*)(slots + TILE_M);       // [0]=sum(s*w) [1]=sum(b*w)

  int e = blockIdx.x >> 7;
  int t = blockIdx.x & (TILES_PER_E - 1);
  int beg = offsets[e] + t * TILE_M;
  int end = offsets[e + 1];
  if (beg >= end) return;
  int m_local = min(TILE_M, end - beg);

  int tid = threadIdx.x;
  const float*  W1g  = W1  + (size_t)e * D * D;
  const float*  W2g  = W2  + (size_t)e * D * D;
  const __bf16* Wt1g = Wt1 + (size_t)e * D * D;
  const __bf16* Wt2g = Wt2 + (size_t)e * D * D;

  if (tid < 2) red[tid] = 0.f;
  if (tid < TILE_M) slots[tid] = (tid < m_local) ? slotlist[beg + tid] : -1;
  __syncthreads();

  {   // fold LN+conf head: scw[d] = scale*w ; red = {sum scale*w, sum bias*w}
    float a = 0.f, b = 0.f;
    for (int d = tid; d < D; d += 256) {
      float w  = cw[e * D + d];
      float sc = lns[e * D + d] * w;
      scw[d] = sc;
      a += sc;
      b += lnb[e * D + d] * w;
    }
    atomicAdd(&red[0], a);
    atomicAdd(&red[1], b);
  }

  // stage gathered x rows as bf16 into LDS
  for (int i = tid; i < TILE_M * D; i += 256) {
    int r = i >> 10, c = i & (D - 1);
    int slot = slots[r];
    float v = (slot >= 0) ? x[(size_t)(slot >> 1) * D + c] : 0.f;
    xb[i] = (__bf16)v;
  }
  __syncthreads();

  int wave = tid >> 5, lane = tid & 31;
  int lr = lane & 15, lhalf = lane >> 4;

  // ---------------- GEMM1: h = gelu(x @ W1 + b1) -> LDS bf16
  for (int sp = wave; sp < D / 32; sp += 8) {   // 2 column strips per iter
    int col0 = sp * 32 + lr;
    int col1 = col0 + 16;
    v8f acc[8] = {};
    for (int kb = 0; kb < D; kb += 32) {
      v16bf bf0, bf1;
      if constexpr (PRE) {
        bf0 = load_bfragB_bt(Wt1g, kb, col0, lhalf);
        bf1 = load_bfragB_bt(Wt1g, kb, col1, lhalf);
        __builtin_prefetch(Wt1g + (size_t)col0 * D + kb + 32, 0, 1);
        __builtin_prefetch(Wt1g + (size_t)col1 * D + kb + 32, 0, 1);
      } else {
        bf0 = load_bfragB_f32(W1g, kb, col0, lhalf);
        bf1 = load_bfragB_f32(W1g, kb, col1, lhalf);
        __builtin_prefetch(&W1g[(size_t)(kb + 32 + lhalf * 16) * D + col0], 0, 1);
      }
      int kb0 = kb + lhalf * 16;
#pragma unroll
      for (int rt = 0; rt < 4; ++rt) {
        v16bf a = load_afragLDS(xb + (rt * 16 + lr) * D + kb0);
        acc[rt]     = __builtin_amdgcn_wmma_f32_16x16x32_bf16(
            false, a, false, bf0, (short)0, acc[rt],     false, false);
        acc[rt + 4] = __builtin_amdgcn_wmma_f32_16x16x32_bf16(
            false, a, false, bf1, (short)0, acc[rt + 4], false, false);
      }
    }
    float bias0 = b1[e * D + col0];
    float bias1 = b1[e * D + col1];
#pragma unroll
    for (int rt = 0; rt < 4; ++rt)
#pragma unroll
      for (int r = 0; r < 8; ++r) {
        int row = rt * 16 + lhalf * 8 + r;
        hb[row * D + col0] = (__bf16)gelu_exact(acc[rt][r]     + bias0);
        hb[row * D + col1] = (__bf16)gelu_exact(acc[rt + 4][r] + bias1);
      }
  }
  __syncthreads();

  // ---------------- GEMM2: y = h @ W2 + b2 -> scatter to moe_out
  for (int sp = wave; sp < D / 32; sp += 8) {
    int col0 = sp * 32 + lr;
    int col1 = col0 + 16;
    v8f acc[8] = {};
    for (int kb = 0; kb < D; kb += 32) {
      v16bf bf0, bf1;
      if constexpr (PRE) {
        bf0 = load_bfragB_bt(Wt2g, kb, col0, lhalf);
        bf1 = load_bfragB_bt(Wt2g, kb, col1, lhalf);
        __builtin_prefetch(Wt2g + (size_t)col0 * D + kb + 32, 0, 1);
        __builtin_prefetch(Wt2g + (size_t)col1 * D + kb + 32, 0, 1);
      } else {
        bf0 = load_bfragB_f32(W2g, kb, col0, lhalf);
        bf1 = load_bfragB_f32(W2g, kb, col1, lhalf);
        __builtin_prefetch(&W2g[(size_t)(kb + 32 + lhalf * 16) * D + col0], 0, 1);
      }
      int kb0 = kb + lhalf * 16;
#pragma unroll
      for (int rt = 0; rt < 4; ++rt) {
        v16bf a = load_afragLDS(hb + (rt * 16 + lr) * D + kb0);
        acc[rt]     = __builtin_amdgcn_wmma_f32_16x16x32_bf16(
            false, a, false, bf0, (short)0, acc[rt],     false, false);
        acc[rt + 4] = __builtin_amdgcn_wmma_f32_16x16x32_bf16(
            false, a, false, bf1, (short)0, acc[rt + 4], false, false);
      }
    }
    float bias0 = b2[e * D + col0];
    float bias1 = b2[e * D + col1];
#pragma unroll
    for (int rt = 0; rt < 4; ++rt)
#pragma unroll
      for (int r = 0; r < 8; ++r) {
        int row = rt * 16 + lhalf * 8 + r;
        if (row < m_local) {
          size_t base = (size_t)slots[row] * D;
          moe_out[base + col0] = acc[rt][r]     + bias0;
          moe_out[base + col1] = acc[rt + 4][r] + bias1;
        }
      }
  }
  __threadfence();
  __syncthreads();

  // ---------------- row stats -> conf = sigmoid(LN(y) @ cw + cb)
  float S_scw = red[0], B_bcw = red[1];
  for (int r = wave; r < TILE_M; r += 8) {
    if (r >= m_local) continue;
    size_t base = (size_t)slots[r] * D;
    float s1 = 0.f, s2 = 0.f, s3 = 0.f;
    for (int c = lane; c < D; c += 32) {
      float v = moe_out[base + c];
      s1 += v; s2 += v * v; s3 += v * scw[c];
    }
#pragma unroll
    for (int off = 16; off > 0; off >>= 1) {
      s1 += __shfl_xor(s1, off, 32);
      s2 += __shfl_xor(s2, off, 32);
      s3 += __shfl_xor(s3, off, 32);
    }
    if (lane == 0) {
      float mu  = s1 * (1.0f / D);
      float var = s2 * (1.0f / D) - mu * mu;
      float rs  = rsqrtf(var + LN_EPS);
      float logit = rs * (s3 - mu * S_scw) + B_bcw + cb[e];
      conf_out[slots[r]] = 1.0f / (1.0f + expf(-logit));
    }
  }
}

// ---------------------------------------------------------------- softmax head
__global__ void prob_kernel(const float* __restrict__ moe_out,
                            const float* __restrict__ pw,
                            const float* __restrict__ pb,
                            float* __restrict__ prob) {
  int wave = threadIdx.x >> 5, lane = threadIdx.x & 31;
  int row = blockIdx.x * 8 + wave;
  if (row >= N_TOK * TOPK) return;
  size_t base = (size_t)row * D;
  float d0 = 0.f, d1 = 0.f;
  for (int c = lane; c < D; c += 32) {
    float v = moe_out[base + c];
    d0 += v * pw[c * NLAB + 0];
    d1 += v * pw[c * NLAB + 1];
  }
#pragma unroll
  for (int off = 16; off > 0; off >>= 1) {
    d0 += __shfl_xor(d0, off, 32);
    d1 += __shfl_xor(d1, off, 32);
  }
  if (lane == 0) {
    float z0 = d0 + pb[0], z1 = d1 + pb[1];
    float m  = fmaxf(z0, z1);
    float e0 = expf(z0 - m), e1 = expf(z1 - m);
    float inv = 1.0f / (e0 + e1);
    prob[row * NLAB + 0] = e0 * inv;
    prob[row * NLAB + 1] = e1 * inv;
  }
}

// ---------------------------------------------------------------- launcher
extern "C" void kernel_launch(void* const* d_in, const int* in_sizes, int n_in,
                              void* d_out, int out_size, void* d_ws, size_t ws_size,
                              hipStream_t stream) {
  const float* x   = (const float*)d_in[0];
  const float* gw  = (const float*)d_in[1];
  const float* gb  = (const float*)d_in[2];
  const float* W1  = (const float*)d_in[3];
  const float* b1  = (const float*)d_in[4];
  const float* W2  = (const float*)d_in[5];
  const float* b2  = (const float*)d_in[6];
  const float* lns = (const float*)d_in[7];
  const float* lnb = (const float*)d_in[8];
  const float* cw  = (const float*)d_in[9];
  const float* cb  = (const float*)d_in[10];
  const float* pw  = (const float*)d_in[11];
  const float* pb  = (const float*)d_in[12];

  float* moe  = (float*)d_out;
  float* conf = moe + (size_t)N_TOK * TOPK * D;
  float* prob = conf + (size_t)N_TOK * TOPK;

  int* counts   = (int*)d_ws;            // 16
  int* cursor   = counts + 16;           // 16
  int* offsets  = cursor + 16;           // 17
  int* idx      = (int*)d_ws + 64;       // N*K
  int* slotlist = idx + N_TOK * TOPK;    // N*K

  const size_t wt_off = 143360;                     // past control block
  const size_t wt_elems = (size_t)E * D * D;        // per matrix
  bool pre = ws_size >= wt_off + 2 * wt_elems * sizeof(__bf16);
  __bf16* Wt1 = (__bf16*)((char*)d_ws + wt_off);
  __bf16* Wt2 = Wt1 + wt_elems;

  hipMemsetAsync(counts, 0, E * sizeof(int), stream);
  gate_topk_kernel<<<N_TOK / 16, 256, 0, stream>>>(x, gw, gb, idx, counts);
  scan_kernel<<<1, 32, 0, stream>>>(counts, offsets, cursor);
  scatter_kernel<<<N_TOK / 256, 256, 0, stream>>>(idx, offsets, cursor, slotlist);

  if (pre) {
    dim3 g(D / 32, D / 32, E), b(32, 8);
    transpose_cvt_kernel<<<g, b, 0, stream>>>(W1, Wt1);
    transpose_cvt_kernel<<<g, b, 0, stream>>>(W2, Wt2);
  }

  size_t smem = (size_t)2 * TILE_M * D * sizeof(__bf16)   // xb + hb
              + D * sizeof(float) + TILE_M * sizeof(int) + 16;
  hipFuncSetAttribute(reinterpret_cast<const void*>(&expert_kernel<true>),
                      hipFuncAttributeMaxDynamicSharedMemorySize, (int)smem);
  hipFuncSetAttribute(reinterpret_cast<const void*>(&expert_kernel<false>),
                      hipFuncAttributeMaxDynamicSharedMemorySize, (int)smem);

  if (pre)
    expert_kernel<true><<<E * TILES_PER_E, 256, smem, stream>>>(
        x, W1, b1, W2, b2, Wt1, Wt2, lns, lnb, cw, cb, offsets, slotlist, moe, conf);
  else
    expert_kernel<false><<<E * TILES_PER_E, 256, smem, stream>>>(
        x, W1, b1, W2, b2, Wt1, Wt2, lns, lnb, cw, cb, offsets, slotlist, moe, conf);

  prob_kernel<<<(N_TOK * TOPK) / 8, 256, 0, stream>>>(moe, pw, pb, prob);
}
